// RBF_32100585570665
// MI455X (gfx1250) — matrix operand order repaired and checked by reading, not compiled, not run on hardware
//
#include <hip/hip_runtime.h>

typedef unsigned int uint_t;
typedef unsigned short ushort_t;

typedef __bf16 v16bf __attribute__((ext_vector_type(16)));
typedef float  v8f   __attribute__((ext_vector_type(8)));
typedef uint_t v8u   __attribute__((ext_vector_type(8)));

// ---------------------------------------------------------------------------
// Tile configuration: 128x128 output tile per 256-thread block (8 wave32s),
// K chunked by 32 (bf16 WMMA K). LDS row stride 40 bf16 (80B): 16B aligned
// for b128 reads, and 20-dword stride -> 16 distinct banks across 16 lanes.
// Software pipeline: chunk k+1 fp32 data staged in VGPRs while chunk k's
// WMMAs execute.
// ---------------------------------------------------------------------------
#define BM   128
#define BN   128
#define BK   32
#define LDT  40   // padded LDS row stride in bf16 elements

// Split an fp32 value into bf16 hi (RNE) + bf16 lo (RNE of residual).
__device__ __forceinline__ void bf16_split(float f, ushort_t& h, ushort_t& l) {
    uint_t u  = __float_as_uint(f);
    uint_t hr = (u + 0x7FFFu + ((u >> 16) & 1u)) & 0xFFFF0000u;
    float  hf = __uint_as_float(hr);
    float  lo = f - hf;
    uint_t ul = __float_as_uint(lo);
    uint_t lr = ul + 0x7FFFu + ((ul >> 16) & 1u);
    h = (ushort_t)(hr >> 16);
    l = (ushort_t)(lr >> 16);
}

__device__ __forceinline__ void cvt4(float4 v, uint_t& h01, uint_t& h23,
                                     uint_t& l01, uint_t& l23) {
    ushort_t h0, h1, h2, h3, l0, l1, l2, l3;
    bf16_split(v.x, h0, l0);
    bf16_split(v.y, h1, l1);
    bf16_split(v.z, h2, l2);
    bf16_split(v.w, h3, l3);
    h01 = (uint_t)h0 | ((uint_t)h1 << 16);
    h23 = (uint_t)h2 | ((uint_t)h3 << 16);
    l01 = (uint_t)l0 | ((uint_t)l1 << 16);
    l23 = (uint_t)l2 | ((uint_t)l3 << 16);
}

// Load a 16-bf16 fragment from two 16B-aligned LDS chunks (ds_load_b128 x2).
__device__ __forceinline__ v16bf load_frag2(const ushort_t* p0, const ushort_t* p1) {
    uint4 q0 = *reinterpret_cast<const uint4*>(p0);
    uint4 q1 = *reinterpret_cast<const uint4*>(p1);
    v8u t;
    t[0] = q0.x; t[1] = q0.y; t[2] = q0.z; t[3] = q0.w;
    t[4] = q1.x; t[5] = q1.y; t[6] = q1.z; t[7] = q1.w;
    return __builtin_bit_cast(v16bf, t);
}

// ---------------------------------------------------------------------------
// Kernel 1: per-row squared L2 norms (wave per row, wave32 shuffle reduction)
// ---------------------------------------------------------------------------
__global__ __launch_bounds__(256)
void rowsq_kernel(const float* __restrict__ src, float* __restrict__ dst,
                  int rows, int cols) {
    int lane = threadIdx.x & 31;
    int wv   = threadIdx.x >> 5;
    int row  = blockIdx.x * 8 + wv;
    if (row >= rows) return;
    const float* p = src + (size_t)row * cols;
    float s = 0.0f;
    for (int i = lane; i < cols; i += 32) {
        float v = p[i];
        s = fmaf(v, v, s);
    }
    #pragma unroll
    for (int off = 16; off > 0; off >>= 1)
        s += __shfl_xor(s, off, 32);
    if (lane == 0) dst[row] = s;
}

// ---------------------------------------------------------------------------
// Kernel 2: fused split-bf16 WMMA GEMM + RBF epilogue.
//   cross = x . c  via (hi*hi + hi*lo + lo*hi) bf16 WMMAs (~fp32 accuracy)
//   out   = exp(-max(x2 + c2 - 2*cross, 0) * exp(-2*log_sigma))
// ---------------------------------------------------------------------------
__global__ __launch_bounds__(256)
void rbf_wmma_kernel(const float* __restrict__ x,
                     const float* __restrict__ cen,
                     const float* __restrict__ ls,
                     const float* __restrict__ xsq,
                     const float* __restrict__ csq,
                     float* __restrict__ out,
                     int N, int C, int D) {
    __shared__ ushort_t sAh[BM * LDT];
    __shared__ ushort_t sAl[BM * LDT];
    __shared__ ushort_t sBh[BN * LDT];
    __shared__ ushort_t sBl[BN * LDT];

    const int tid  = threadIdx.x;
    const int lane = tid & 31;
    const int wv   = tid >> 5;       // 0..7
    const int wr   = wv >> 1;        // 0..3 : 32-row band
    const int wc   = wv & 1;         // 0..1 : 64-col band
    const int lhalf = lane >> 4;     // K-half selector
    const int lmod  = lane & 15;

    const int bm0 = blockIdx.x * BM;
    const int bn0 = blockIdx.y * BN;

    // Loader mapping: thread t -> row t>>1, 16 consecutive cols at (t&1)*16
    const int lr = tid >> 1;
    const int lc = (tid & 1) * 16;
    const float* gA = x   + (size_t)(bm0 + lr) * D + lc;
    const float* gB = cen + (size_t)(bn0 + lr) * D + lc;
    uint_t* dAh = (uint_t*)(sAh + lr * LDT + lc);
    uint_t* dAl = (uint_t*)(sAl + lr * LDT + lc);
    uint_t* dBh = (uint_t*)(sBh + lr * LDT + lc);
    uint_t* dBl = (uint_t*)(sBl + lr * LDT + lc);

    v8f acc[2][4] = {};

    const int nk = D / BK;

    // --- software pipeline: stage chunk 0 in registers ---------------------
    float4 ra[4], rb[4];
    #pragma unroll
    for (int q = 0; q < 4; ++q) {
        ra[q] = reinterpret_cast<const float4*>(gA)[q];
        rb[q] = reinterpret_cast<const float4*>(gB)[q];
    }

    for (int kt = 0; kt < nk; ++kt) {
        // Convert staged fp32 -> bf16 hi/lo and commit to LDS.
        #pragma unroll
        for (int q = 0; q < 4; ++q) {
            uint_t h01, h23, l01, l23;
            cvt4(ra[q], h01, h23, l01, l23);
            dAh[2 * q] = h01; dAh[2 * q + 1] = h23;
            dAl[2 * q] = l01; dAl[2 * q + 1] = l23;
            cvt4(rb[q], h01, h23, l01, l23);
            dBh[2 * q] = h01; dBh[2 * q + 1] = h23;
            dBl[2 * q] = l01; dBl[2 * q + 1] = l23;
        }
        __syncthreads();

        // Issue next chunk's global loads now; they overlap the WMMA block.
        if (kt + 1 < nk) {
            const float* pa = gA + (kt + 1) * BK;
            const float* pb = gB + (kt + 1) * BK;
            #pragma unroll
            for (int q = 0; q < 4; ++q) {
                ra[q] = reinterpret_cast<const float4*>(pa)[q];
                rb[q] = reinterpret_cast<const float4*>(pb)[q];
            }
            if (kt + 2 < nk) {           // warm L2 one chunk further ahead
                __builtin_prefetch(pa + BK, 0, 3);
                __builtin_prefetch(pb + BK, 0, 3);
            }
        }

        // A fragments (16x32 bf16): lane<16 -> K 0..7 / 16..23, lane>=16 ->
        // K 8..15 / 24..31 (ISA 7.12.2 16-bit A layout).
        v16bf ah[2], al[2];
        #pragma unroll
        for (int rm = 0; rm < 2; ++rm) {
            int arow = wr * 32 + rm * 16 + lmod;
            const ushort_t* bh = sAh + arow * LDT;
            const ushort_t* bl = sAl + arow * LDT;
            ah[rm] = load_frag2(bh + lhalf * 8, bh + 16 + lhalf * 8);
            al[rm] = load_frag2(bl + lhalf * 8, bl + 16 + lhalf * 8);
        }

        #pragma unroll
        for (int cn = 0; cn < 4; ++cn) {
            // B fragment (32x16 bf16): lanes 0-15 hold K 0..7 / 16..23, lanes 16-31
            // hold K 8..15 / 24..31 (same K order as A).
            int brow = wc * 64 + cn * 16 + lmod;
            const ushort_t* pbh = sBh + brow * LDT + lhalf * 8;
            const ushort_t* pbl = sBl + brow * LDT + lhalf * 8;
            v16bf bh = load_frag2(pbh, pbh + 16);
            v16bf bl = load_frag2(pbl, pbl + 16);
            #pragma unroll
            for (int rm = 0; rm < 2; ++rm) {
                acc[rm][cn] = __builtin_amdgcn_wmma_f32_16x16x32_bf16(
                    false, ah[rm], false, bh, (short)0, acc[rm][cn], false, false);
                acc[rm][cn] = __builtin_amdgcn_wmma_f32_16x16x32_bf16(
                    false, ah[rm], false, bl, (short)0, acc[rm][cn], false, false);
                acc[rm][cn] = __builtin_amdgcn_wmma_f32_16x16x32_bf16(
                    false, al[rm], false, bh, (short)0, acc[rm][cn], false, false);
            }
        }
        __syncthreads();
    }

    // Epilogue: C/D layout -> VGPR r holds (M = r + 8*lhalf, N = lmod).
    #pragma unroll
    for (int cn = 0; cn < 4; ++cn) {
        int col     = bn0 + wc * 64 + cn * 16 + lmod;
        float csqv  = csq[col];
        float inv2  = __expf(-2.0f * ls[col]);
        #pragma unroll
        for (int rm = 0; rm < 2; ++rm) {
            #pragma unroll
            for (int r = 0; r < 8; ++r) {
                int row  = bm0 + wr * 32 + rm * 16 + r + 8 * lhalf;
                float sq = fmaxf(xsq[row] + csqv - 2.0f * acc[rm][cn][r], 0.0f);
                out[(size_t)row * C + col] = __expf(-sq * inv2);
            }
        }
    }
}

// ---------------------------------------------------------------------------
extern "C" void kernel_launch(void* const* d_in, const int* in_sizes, int n_in,
                              void* d_out, int out_size, void* d_ws, size_t ws_size,
                              hipStream_t stream) {
    const float* x   = (const float*)d_in[0];   // (N, D) fp32
    const float* cen = (const float*)d_in[1];   // (C, D) fp32
    const float* ls  = (const float*)d_in[2];   // (C,)   fp32
    float* out = (float*)d_out;                 // (N, C) fp32

    const int C = in_sizes[2];
    const int D = in_sizes[1] / C;
    const int N = in_sizes[0] / D;

    float* xsq = (float*)d_ws;       // N floats
    float* csq = xsq + N;            // C floats

    rowsq_kernel<<<(N + 7) / 8, 256, 0, stream>>>(x, xsq, N, D);
    rowsq_kernel<<<(C + 7) / 8, 256, 0, stream>>>(cen, csq, C, D);

    dim3 grid(N / BM, C / BN);
    rbf_wmma_kernel<<<grid, 256, 0, stream>>>(x, cen, ls, xsq, csq, out, N, C, D);
}
